// GIN_28716151341446
// MI455X (gfx1250) — matrix operand order repaired and checked by reading, not compiled, not run on hardware
//
#include <hip/hip_runtime.h>
#include <hip/hip_bf16.h>

// ---------------------------------------------------------------------------
// Types for CDNA5 WMMA (wave32)
// ---------------------------------------------------------------------------
typedef __attribute__((ext_vector_type(16))) __bf16 v16bf;
typedef __attribute__((ext_vector_type(8)))  __bf16 v8bf;
typedef __attribute__((ext_vector_type(8)))  float  v8f;

#define DIM 256
#define NGRAPH 500
#define GN_EPS 1e-6f

// ---------------------------------------------------------------------------
// Utility kernels
// ---------------------------------------------------------------------------
__global__ void fill_zero_k(float* __restrict__ p, long n) {
    long i = (long)blockIdx.x * blockDim.x + threadIdx.x;
    if (i < n) p[i] = 0.0f;
}

__global__ void accum_counts_k(const int* __restrict__ gid, float* __restrict__ counts, int N) {
    int i = blockIdx.x * blockDim.x + threadIdx.x;
    if (i < N) atomicAdd(&counts[gid[i]], 1.0f);
}

__global__ void f32_to_bf16_k(const float* __restrict__ in, __bf16* __restrict__ out, long n) {
    long i = (long)blockIdx.x * blockDim.x + threadIdx.x;
    if (i < n) out[i] = (__bf16)in[i];
}

// ---------------------------------------------------------------------------
// Edge aggregation: agg[dst] += h[src]   (agg pre-initialized to h => rst)
// One block handles 8 edges; threadIdx.x = feature 0..255 (coalesced 1KB rows)
// ---------------------------------------------------------------------------
#define EDGES_PER_BLOCK 8
__global__ __launch_bounds__(DIM) void edge_scatter_k(
    const float* __restrict__ h, const int* __restrict__ src,
    const int* __restrict__ dst, float* __restrict__ agg, int E) {
    int d = threadIdx.x;
    int e0 = blockIdx.x * EDGES_PER_BLOCK;
#pragma unroll
    for (int i = 0; i < EDGES_PER_BLOCK; ++i) {
        int e = e0 + i;
        if (e < E) {
            int s = src[e];
            int t = dst[e];
            float v = h[(long)s * DIM + d];
            atomicAdd(&agg[(long)t * DIM + d], v);
        }
    }
}

// ---------------------------------------------------------------------------
// Pre-pack fp32 weights (W[k][n], k=input dim) into WMMA B-matrix bf16 layout.
// Bp element index = (((mat*16+nt)*8+ks)*32+lane)*16 + e
//   n = nt*16 + (lane&15);  k = ks*32 + (lane>>4)*16 + e
// ---------------------------------------------------------------------------
__global__ void prepack_w_k(const float* __restrict__ W1, const float* __restrict__ W2,
                            __bf16* __restrict__ Bp, long total) {
    long idx = (long)blockIdx.x * blockDim.x + threadIdx.x;
    if (idx >= total) return;
    int m    = (int)(idx >> 16);         // matrix (layer*2 + which), 65536 elems each
    int r    = (int)(idx & 65535);
    int nt   = r >> 12;                  // / (8*32*16)
    int r2   = r & 4095;
    int ks   = r2 >> 9;                  // / (32*16)
    int r3   = r2 & 511;
    int lane = r3 >> 4;
    int e    = r3 & 15;
    int n    = nt * 16 + (lane & 15);
    int k    = ks * 32 + (lane >> 4) * 16 + e;
    int layer = m >> 1;
    const float* W = (m & 1) ? W2 : W1;
    float v = W[(long)layer * DIM * DIM + (long)k * DIM + n];
    Bp[idx] = (__bf16)v;
}

// ---------------------------------------------------------------------------
// WMMA GEMM: Z[N,256] = A[N,256](bf16) @ Bp(packed 256x256 bf16) + bias
// Fused epilogue: atomicAdd per-graph mean accumulator.
// Block = 8 waves: 4 waves in M (64 rows), 2 waves in N (2 x 8 tiles of 16).
// ---------------------------------------------------------------------------
__global__ __launch_bounds__(256) void gemm_bias_meansum_k(
    const __bf16* __restrict__ A, const __bf16* __restrict__ Bp,
    const float* __restrict__ bias, const int* __restrict__ gid,
    float* __restrict__ Z, float* __restrict__ meanAcc, int N) {
    int wave = threadIdx.x >> 5;
    int lane = threadIdx.x & 31;
    int wm = wave & 3;      // M-wave: 0..3
    int wn = wave >> 2;     // N-wave: 0..1
    int rowBase = blockIdx.x * 64 + wm * 16;
    int lrow = rowBase + (lane & 15);
    int arow = (lrow < N) ? lrow : (N - 1);
    int kh = lane >> 4;     // K half select (lanes 16-31 hold shifted K)
    const __bf16* aptr = A + (long)arow * DIM;

    v8f acc[8] = {};
#pragma unroll
    for (int ks = 0; ks < 8; ++ks) {
        int k0 = ks * 32;
        // A lane layout: lanes 0-15: K = k0+{0..15}
        //                lanes 16-31: K = k0+16+{0..15}
        v8bf lo = *(const v8bf*)(aptr + k0 + kh * 16);
        v8bf hi = *(const v8bf*)(aptr + k0 + kh * 16 + 8);
        v16bf a = __builtin_shufflevector(lo, hi, 0, 1, 2, 3, 4, 5, 6, 7,
                                          8, 9, 10, 11, 12, 13, 14, 15);
#pragma unroll
        for (int t = 0; t < 8; ++t) {
            int nt = wn * 8 + t;
            const v16bf b = *(const v16bf*)(Bp + ((long)((nt * 8 + ks) * 32 + lane)) * 16);
            acc[t] = __builtin_amdgcn_wmma_f32_16x16x32_bf16(
                false, a, false, b, (short)0, acc[t], false, false);
        }
    }
    // Epilogue: C/D layout -> VGPR r: lanes 0-15 = M r, lanes 16-31 = M r+8.
#pragma unroll
    for (int t = 0; t < 8; ++t) {
        int col = (wn * 8 + t) * 16 + (lane & 15);
        float bv = bias[col];
#pragma unroll
        for (int r = 0; r < 8; ++r) {
            int row = rowBase + r + kh * 8;
            if (row < N) {
                float v = acc[t][r] + bv;
                Z[(long)row * DIM + col] = v;
                int g = gid[row];
                atomicAdd(&meanAcc[g * DIM + col], v);
            }
        }
    }
}

// ---------------------------------------------------------------------------
// GraphNorm helpers
// ---------------------------------------------------------------------------
__global__ void mean_div_k(float* __restrict__ mean, const float* __restrict__ counts, int GD) {
    int i = blockIdx.x * blockDim.x + threadIdx.x;
    if (i < GD) mean[i] /= fmaxf(counts[i >> 8], 1.0f);
}

__global__ void gn_sub_var_k(float* __restrict__ z, const float* __restrict__ mean,
                             const int* __restrict__ gid, const float* __restrict__ ms,
                             float* __restrict__ varAcc, long NE) {
    long idx = (long)blockIdx.x * blockDim.x + threadIdx.x;
    if (idx >= NE) return;
    int row = (int)(idx >> 8);
    int col = (int)(idx & 255);
    int g = gid[row];
    float s = z[idx] - mean[g * DIM + col] * ms[col];
    z[idx] = s;
    atomicAdd(&varAcc[g * DIM + col], s * s);
}

__global__ void var_finalize_k(float* __restrict__ var, const float* __restrict__ counts, int GD) {
    int i = blockIdx.x * blockDim.x + threadIdx.x;
    if (i < GD) var[i] = rsqrtf(var[i] / fmaxf(counts[i >> 8], 1.0f) + GN_EPS);
}

// affine + relu, emit bf16 (feeds next GEMM)
__global__ void gn_apply_bf16_k(const float* __restrict__ z, const float* __restrict__ inv,
                                const int* __restrict__ gid, const float* __restrict__ w,
                                const float* __restrict__ b, __bf16* __restrict__ out, long NE) {
    long idx = (long)blockIdx.x * blockDim.x + threadIdx.x;
    if (idx >= NE) return;
    int row = (int)(idx >> 8);
    int col = (int)(idx & 255);
    int g = gid[row];
    float v = w[col] * z[idx] * inv[g * DIM + col] + b[col];
    out[idx] = (__bf16)fmaxf(v, 0.0f);
}

// affine + relu + optional residual, emit fp32 h_next
__global__ void gn_apply_res_k(const float* __restrict__ z, const float* __restrict__ inv,
                               const int* __restrict__ gid, const float* __restrict__ w,
                               const float* __restrict__ b, const float* __restrict__ hprev,
                               float* __restrict__ hnext, int useRes, long NE) {
    long idx = (long)blockIdx.x * blockDim.x + threadIdx.x;
    if (idx >= NE) return;
    int row = (int)(idx >> 8);
    int col = (int)(idx & 255);
    int g = gid[row];
    float v = fmaxf(w[col] * z[idx] * inv[g * DIM + col] + b[col], 0.0f);
    hnext[idx] = v + (useRes ? hprev[idx] : 0.0f);
}

// ---------------------------------------------------------------------------
// Mean pooling + final projection
// ---------------------------------------------------------------------------
__global__ void pool_sum_k(const float* __restrict__ h, const int* __restrict__ gid,
                           float* __restrict__ pooled, long NE) {
    long idx = (long)blockIdx.x * blockDim.x + threadIdx.x;
    if (idx >= NE) return;
    int row = (int)(idx >> 8);
    int col = (int)(idx & 255);
    atomicAdd(&pooled[gid[row] * DIM + col], h[idx]);
}

__global__ void final_proj_k(const float* __restrict__ pooled, const float* __restrict__ counts,
                             const float* __restrict__ Wp, const float* __restrict__ bp,
                             float* __restrict__ out, int OUT) {
    int g = blockIdx.x;
    int o = threadIdx.x;
    float c = 1.0f / fmaxf(counts[g], 1.0f);
    float s = 0.0f;
    for (int d = 0; d < DIM; ++d)
        s += pooled[g * DIM + d] * Wp[d * OUT + o];
    out[g * OUT + o] = s * c + bp[o];
}

// ---------------------------------------------------------------------------
// Host orchestration
// ---------------------------------------------------------------------------
extern "C" void kernel_launch(void* const* d_in, const int* in_sizes, int n_in,
                              void* d_out, int out_size, void* d_ws, size_t ws_size,
                              hipStream_t stream) {
    const float* h_in = (const float*)d_in[0];
    const float* W1   = (const float*)d_in[1];
    const float* b1   = (const float*)d_in[2];
    const float* W2   = (const float*)d_in[3];
    const float* b2   = (const float*)d_in[4];
    const float* n1w  = (const float*)d_in[5];
    const float* n1b  = (const float*)d_in[6];
    const float* n1m  = (const float*)d_in[7];
    const float* n2w  = (const float*)d_in[8];
    const float* n2b  = (const float*)d_in[9];
    const float* n2m  = (const float*)d_in[10];
    const float* Wp   = (const float*)d_in[11];
    const float* bp   = (const float*)d_in[12];
    const int* esrc   = (const int*)d_in[13];
    const int* edst   = (const int*)d_in[14];
    const int* gid    = (const int*)d_in[15];

    const int N   = in_sizes[0] / DIM;
    const int L   = in_sizes[1] / (DIM * DIM);
    const int E   = in_sizes[13];
    const int OUT = in_sizes[12];
    const int G   = NGRAPH;
    float* out = (float*)d_out;

    // workspace carve (256B aligned)
    char* ws = (char*)d_ws;
    auto carve = [&](size_t bytes) -> char* {
        char* p = ws;
        ws += (bytes + 255) & ~(size_t)255;
        return p;
    };
    const long NE = (long)N * DIM;
    const size_t NDf = (size_t)NE * sizeof(float);
    float*  hA     = (float*)carve(NDf);
    float*  hB     = (float*)carve(NDf);
    float*  agg    = (float*)carve(NDf);
    float*  zbuf   = (float*)carve(NDf);
    __bf16* rstb   = (__bf16*)carve(NDf / 2);
    __bf16* zb     = (__bf16*)carve(NDf / 2);
    float*  meanb  = (float*)carve((size_t)G * DIM * sizeof(float));
    float*  varb   = (float*)carve((size_t)G * DIM * sizeof(float));
    float*  counts = (float*)carve((size_t)G * sizeof(float));
    float*  pooled = (float*)carve((size_t)G * DIM * sizeof(float));
    __bf16* Bp     = (__bf16*)carve((size_t)L * 2 * DIM * DIM * sizeof(__bf16));
    (void)ws_size; (void)n_in; (void)out_size;

    const int TB = 256;
    const int gGD = (G * DIM + TB - 1) / TB;
    const int gNE = (int)((NE + TB - 1) / TB);

    // 1) pre-pack weights into WMMA-B bf16 layout
    {
        long total = (long)L * 2 * DIM * DIM;
        prepack_w_k<<<(int)((total + TB - 1) / TB), TB, 0, stream>>>(W1, W2, Bp, total);
    }
    // 2) per-graph node counts
    fill_zero_k<<<(G + TB - 1) / TB, TB, 0, stream>>>(counts, G);
    accum_counts_k<<<(N + TB - 1) / TB, TB, 0, stream>>>(gid, counts, N);

    const float* hcur = h_in;
    for (int i = 0; i < L; ++i) {
        float* hnext = (i & 1) ? hB : hA;
        // GIN aggregation: rst = h + sum_{src->dst} h[src]
        hipMemcpyAsync(agg, hcur, NDf, hipMemcpyDeviceToDevice, stream);
        edge_scatter_k<<<(E + EDGES_PER_BLOCK - 1) / EDGES_PER_BLOCK, DIM, 0, stream>>>(
            hcur, esrc, edst, agg, E);
        f32_to_bf16_k<<<gNE, TB, 0, stream>>>(agg, rstb, NE);

        // MLP layer 1: GEMM(+bias, +mean accum) -> GraphNorm -> ReLU (bf16 out)
        fill_zero_k<<<gGD, TB, 0, stream>>>(meanb, G * DIM);
        gemm_bias_meansum_k<<<(N + 63) / 64, 256, 0, stream>>>(
            rstb, Bp + (size_t)(i * 2 + 0) * DIM * DIM, b1 + i * DIM, gid, zbuf, meanb, N);
        mean_div_k<<<gGD, TB, 0, stream>>>(meanb, counts, G * DIM);
        fill_zero_k<<<gGD, TB, 0, stream>>>(varb, G * DIM);
        gn_sub_var_k<<<gNE, TB, 0, stream>>>(zbuf, meanb, gid, n1m + i * DIM, varb, NE);
        var_finalize_k<<<gGD, TB, 0, stream>>>(varb, counts, G * DIM);
        gn_apply_bf16_k<<<gNE, TB, 0, stream>>>(zbuf, varb, gid, n1w + i * DIM, n1b + i * DIM, zb, NE);

        // MLP layer 2: GEMM -> GraphNorm -> ReLU -> (+residual) -> h_next (f32)
        fill_zero_k<<<gGD, TB, 0, stream>>>(meanb, G * DIM);
        gemm_bias_meansum_k<<<(N + 63) / 64, 256, 0, stream>>>(
            zb, Bp + (size_t)(i * 2 + 1) * DIM * DIM, b2 + i * DIM, gid, zbuf, meanb, N);
        mean_div_k<<<gGD, TB, 0, stream>>>(meanb, counts, G * DIM);
        fill_zero_k<<<gGD, TB, 0, stream>>>(varb, G * DIM);
        gn_sub_var_k<<<gNE, TB, 0, stream>>>(zbuf, meanb, gid, n2m + i * DIM, varb, NE);
        var_finalize_k<<<gGD, TB, 0, stream>>>(varb, counts, G * DIM);
        gn_apply_res_k<<<gNE, TB, 0, stream>>>(zbuf, varb, gid, n2w + i * DIM, n2b + i * DIM,
                                               hcur, hnext, (i != 0) ? 1 : 0, NE);
        hcur = hnext;
    }

    // mean pooling + projection
    fill_zero_k<<<gGD, TB, 0, stream>>>(pooled, G * DIM);
    pool_sum_k<<<gNE, TB, 0, stream>>>(hcur, gid, pooled, NE);
    final_proj_k<<<G, OUT, 0, stream>>>(pooled, counts, Wp, bp, out, OUT);
}
